// HeteroGNN_50199577755961
// MI455X (gfx1250) — matrix operand order repaired and checked by reading, not compiled, not run on hardware
//
#include <hip/hip_runtime.h>

#define HDIM 128
#define WROW 136                 // padded LDS row stride (elems): 272B -> conflict-free b128 reads
#define WTILE (16 * WROW)        // one 16-row weight slice in LDS (elems)

typedef __bf16 bf16x8_t  __attribute__((ext_vector_type(8)));
typedef __bf16 bf16x16_t __attribute__((ext_vector_type(16)));
typedef float  f32x8_t   __attribute__((ext_vector_type(8)));
typedef int    v4i_t     __attribute__((vector_size(4 * sizeof(int))));
typedef __attribute__((address_space(3))) v4i_t lds_v4i_t;

__device__ __forceinline__ float bf2f(unsigned u) {
    return __uint_as_float(u << 16);   // low 16 bits of u are the bf16 pattern
}
__device__ __forceinline__ unsigned short f2bf(float f) {
    unsigned u = __float_as_uint(f);
    u += 0x7fffu + ((u >> 16) & 1u);   // round-to-nearest-even
    return (unsigned short)(u >> 16);
}
__device__ __forceinline__ unsigned pack2bf(float a, float b) {
    return (unsigned)f2bf(a) | ((unsigned)f2bf(b) << 16);
}

// ---------------------------------------------------------------- convert
__global__ __launch_bounds__(256) void k_f32_to_bf16(
    const float* __restrict__ src, unsigned short* __restrict__ dst, int nquads) {
    int i = blockIdx.x * blockDim.x + threadIdx.x;
    if (i >= nquads) return;
    float4 v = reinterpret_cast<const float4*>(src)[i];
    uint2 o;
    o.x = pack2bf(v.x, v.y);
    o.y = pack2bf(v.z, v.w);
    reinterpret_cast<uint2*>(dst)[i] = o;
}

// ------------------------------------------------- fuse conv weights (fp32)
// Wf1 = Wupd[:, :128] @ Wdst ; Wf2 = Wupd[:, 128:] @ Wsrc
// bf  = Wupd @ [bdst; bsrc] + bupd
__global__ __launch_bounds__(HDIM) void k_fuse_weights(
    const float* __restrict__ Wd, const float* __restrict__ bd,
    const float* __restrict__ Ws, const float* __restrict__ bs,
    const float* __restrict__ Wu, const float* __restrict__ bu,
    unsigned short* __restrict__ wf1, unsigned short* __restrict__ wf2,
    float* __restrict__ bf) {
    int o = blockIdx.x, i = threadIdx.x;
    float a1 = 0.f, a2 = 0.f;
    for (int k = 0; k < HDIM; ++k) {
        float u1 = Wu[o * 2 * HDIM + k];
        float u2 = Wu[o * 2 * HDIM + HDIM + k];
        a1 += u1 * Wd[k * HDIM + i];
        a2 += u2 * Ws[k * HDIM + i];
    }
    wf1[o * HDIM + i] = f2bf(a1);
    wf2[o * HDIM + i] = f2bf(a2);
    if (i == 0) {
        float acc = bu[o];
        for (int k = 0; k < HDIM; ++k)
            acc += Wu[o * 2 * HDIM + k] * bd[k] + Wu[o * 2 * HDIM + HDIM + k] * bs[k];
        bf[o] = acc;
    }
}

// ------------------------------------------------- segment scatter-add (mean numerator)
__global__ __launch_bounds__(256) void k_scatter_add(
    const unsigned short* __restrict__ xb,
    const int* __restrict__ srci, const int* __restrict__ dsti,
    float* __restrict__ aggsum, float* __restrict__ counts, int E) {
    int wave = threadIdx.x >> 5, lane = threadIdx.x & 31;
    int e = blockIdx.x * 8 + wave;
    if (e >= E) return;
    int s = srci[e], d = dsti[e];
    uint2 p = reinterpret_cast<const uint2*>(xb + (size_t)s * HDIM + lane * 4)[0];
    float* dp = aggsum + (size_t)d * HDIM + lane * 4;
    atomicAdd(dp + 0, bf2f(p.x));
    atomicAdd(dp + 1, bf2f(p.x >> 16));
    atomicAdd(dp + 2, bf2f(p.y));
    atomicAdd(dp + 3, bf2f(p.y >> 16));
    if (lane == 0) atomicAdd(counts + d, 1.0f);
}

// ------------------------------------------------- mean + pack to bf16
__global__ __launch_bounds__(256) void k_mean_bf16(
    const float* __restrict__ aggsum, const float* __restrict__ counts,
    unsigned short* __restrict__ aggb, int nquads) {
    int i = blockIdx.x * blockDim.x + threadIdx.x;
    if (i >= nquads) return;
    int row = i >> 5;
    float inv = 1.0f / fmaxf(counts[row], 1.0f);
    float4 v = reinterpret_cast<const float4*>(aggsum)[i];
    uint2 o;
    o.x = pack2bf(v.x * inv, v.y * inv);
    o.y = pack2bf(v.z * inv, v.w * inv);
    reinterpret_cast<uint2*>(aggb)[i] = o;
}

// ------------------------------------------------- WMMA fragment loaders
// A 16x32 bf16 (ISA 7.12.2): VGPR0-3 = K[h*8..h*8+7], VGPR4-7 = K[16+h*8..+7]
__device__ __forceinline__ bf16x16_t load_a_frag(
    const unsigned short* __restrict__ rowp, int kbase, int half) {
    bf16x8_t lo = *reinterpret_cast<const bf16x8_t*>(rowp + kbase + half * 8);
    bf16x8_t hi = *reinterpret_cast<const bf16x8_t*>(rowp + kbase + 16 + half * 8);
    return __builtin_shufflevector(lo, hi, 0, 1, 2, 3, 4, 5, 6, 7,
                                   8, 9, 10, 11, 12, 13, 14, 15);
}
// B 32x16 bf16: same K order as A: VGPR0-3 = K[h*8..h*8+7], VGPR4-7 = K[16+h*8..+7]
__device__ __forceinline__ bf16x16_t load_b_frag(
    const unsigned short* rowp, int kbase, int half) {
    bf16x8_t lo = *reinterpret_cast<const bf16x8_t*>(rowp + kbase + half * 8);
    bf16x8_t hi = *reinterpret_cast<const bf16x8_t*>(rowp + kbase + 16 + half * 8);
    return __builtin_shufflevector(lo, hi, 0, 1, 2, 3, 4, 5, 6, 7,
                                   8, 9, 10, 11, 12, 13, 14, 15);
}

// ------------------------------------------------- fused conv GEMM
// out[M,128] (+)= Adst @ W1^T + Aagg @ W2^T + bias
// Block = 8 waves sharing one 16-column weight slice; slice staged in LDS via
// the gfx1250 async global->LDS path (ASYNCcnt) when available.
__global__ __launch_bounds__(256) void k_conv_gemm(
    const unsigned short* __restrict__ Adst,  // [M][128] bf16
    const unsigned short* __restrict__ Aagg,  // [M][128] bf16
    const unsigned short* __restrict__ W1,    // [128][128] bf16 row-major (out,in)
    const unsigned short* __restrict__ W2,
    const float* __restrict__ bias,           // [128]
    float* __restrict__ out, int M, int accumulate) {
    __shared__ unsigned short sW[2 * WTILE];  // 8704 B: W1 slice | W2 slice (padded rows)

    const int tileN = blockIdx.y;
    {
        // stage 16 weight rows of each matrix: 256 threads x 16B per matrix
        const int r = threadIdx.x >> 4;        // 0..15 row in slice
        const int cchunk = threadIdx.x & 15;   // 16 chunks of 8 elems per row
        const unsigned short* g1 =
            W1 + (size_t)(tileN * 16 + r) * HDIM + cchunk * 8;
        const unsigned short* g2 =
            W2 + (size_t)(tileN * 16 + r) * HDIM + cchunk * 8;
        unsigned short* l1 = sW + r * WROW + cchunk * 8;
        unsigned short* l2 = sW + WTILE + r * WROW + cchunk * 8;
#if __has_builtin(__builtin_amdgcn_global_load_async_to_lds_b128)
        __builtin_amdgcn_global_load_async_to_lds_b128(
            (v4i_t*)g1, (lds_v4i_t*)l1, 0, 0);
        __builtin_amdgcn_global_load_async_to_lds_b128(
            (v4i_t*)g2, (lds_v4i_t*)l2, 0, 0);
#if __has_builtin(__builtin_amdgcn_s_wait_asynccnt)
        __builtin_amdgcn_s_wait_asynccnt(0);
#else
        asm volatile("s_wait_asynccnt 0" ::: "memory");
#endif
#else
        *reinterpret_cast<uint4*>(l1) = *reinterpret_cast<const uint4*>(g1);
        *reinterpret_cast<uint4*>(l2) = *reinterpret_cast<const uint4*>(g2);
#endif
    }
    __syncthreads();   // every wave reaches this before any tile guard

    const int wave = threadIdx.x >> 5;
    const int lane = threadIdx.x & 31;
    const int tileM = blockIdx.x * 8 + wave;   // wave-uniform -> EXEC stays all-1
    if (tileM * 16 >= M) return;

    const int half = lane >> 4;
    const int l16 = lane & 15;
    const unsigned short* a1 = Adst + (size_t)(tileM * 16 + l16) * HDIM;
    const unsigned short* a2 = Aagg + (size_t)(tileM * 16 + l16) * HDIM;
    const unsigned short* lw1 = sW + l16 * WROW;           // ds_load_b128, bank-tiled
    const unsigned short* lw2 = sW + WTILE + l16 * WROW;
    const int col = tileN * 16 + l16;

    f32x8_t c = {0.f, 0.f, 0.f, 0.f, 0.f, 0.f, 0.f, 0.f};
#pragma unroll
    for (int k = 0; k < HDIM; k += 32) {
        bf16x16_t a = load_a_frag(a1, k, half);
        bf16x16_t b = load_b_frag(lw1, k, half);
        c = __builtin_amdgcn_wmma_f32_16x16x32_bf16(false, a, false, b,
                                                    (short)0, c, false, false);
    }
#pragma unroll
    for (int k = 0; k < HDIM; k += 32) {
        bf16x16_t a = load_a_frag(a2, k, half);
        bf16x16_t b = load_b_frag(lw2, k, half);
        c = __builtin_amdgcn_wmma_f32_16x16x32_bf16(false, a, false, b,
                                                    (short)0, c, false, false);
    }
    const float bn = bias[col];
    const int mbase = tileM * 16 + half * 8;
    float* op = out + (size_t)mbase * HDIM + col;
    if (accumulate) {
#pragma unroll
        for (int r = 0; r < 8; ++r) op[(size_t)r * HDIM] += c[r] + bn;
    } else {
#pragma unroll
        for (int r = 0; r < 8; ++r) op[(size_t)r * HDIM] = c[r] + bn;
    }
}

// ------------------------------------------------- BatchNorm statistics
__global__ __launch_bounds__(HDIM) void k_bn_stats(
    const float* __restrict__ e, float scale, float* __restrict__ acc, int M) {
    int t = threadIdx.x;
    float s = 0.f, s2 = 0.f;
    for (int r = blockIdx.x; r < M; r += gridDim.x) {
        float v = scale * e[(size_t)r * HDIM + t];
        s += v;
        s2 += v * v;
    }
    atomicAdd(acc + t, s);
    atomicAdd(acc + HDIM + t, s2);
}

__global__ __launch_bounds__(HDIM) void k_bn_finalize(float* __restrict__ st, int M) {
    int t = threadIdx.x;
    float m = st[t] / (float)M;
    float var = st[HDIM + t] / (float)M - m * m;
    st[256 + t] = m;
    st[384 + t] = rsqrtf(var + 1.0f);   // BN eps = 1
}

// ------------------------------------------------- BN apply + LeakyReLU + bf16 pack
__global__ __launch_bounds__(256) void k_bn_apply(
    const float* __restrict__ e, float scale, const float* __restrict__ st,
    const float* __restrict__ g, const float* __restrict__ bsh,
    unsigned short* __restrict__ hb, int nquads) {
    int i = blockIdx.x * blockDim.x + threadIdx.x;
    if (i >= nquads) return;
    int c0 = (i & 31) * 4;
    float4 v = reinterpret_cast<const float4*>(e)[i];
    float vv[4] = {v.x, v.y, v.z, v.w};
    float y[4];
#pragma unroll
    for (int j = 0; j < 4; ++j) {
        int col = c0 + j;
        float t = (scale * vv[j] - st[256 + col]) * st[384 + col] * g[col] + bsh[col];
        y[j] = t >= 0.f ? t : 0.01f * t;
    }
    uint2 o;
    o.x = pack2bf(y[0], y[1]);
    o.y = pack2bf(y[2], y[3]);
    reinterpret_cast<uint2*>(hb)[i] = o;
}

// ------------------------------------------------- link prediction (dot product)
__global__ __launch_bounds__(256) void k_link_dot(
    const unsigned short* __restrict__ h,
    const int* __restrict__ si, const int* __restrict__ di,
    float* __restrict__ out, int EL) {
    int wave = threadIdx.x >> 5, lane = threadIdx.x & 31;
    int e = blockIdx.x * 8 + wave;
    if (e >= EL) return;
    int a = si[e], b = di[e];
    uint2 pa = reinterpret_cast<const uint2*>(h + (size_t)a * HDIM + lane * 4)[0];
    uint2 pb = reinterpret_cast<const uint2*>(h + (size_t)b * HDIM + lane * 4)[0];
    float acc = bf2f(pa.x) * bf2f(pb.x) + bf2f(pa.x >> 16) * bf2f(pb.x >> 16) +
                bf2f(pa.y) * bf2f(pb.y) + bf2f(pa.y >> 16) * bf2f(pb.y >> 16);
#pragma unroll
    for (int off = 16; off > 0; off >>= 1) acc += __shfl_xor(acc, off, 32);
    if (lane == 0) out[e] = acc;
}

// =================================================================
extern "C" void kernel_launch(void* const* d_in, const int* in_sizes, int n_in,
                              void* d_out, int out_size, void* d_ws, size_t ws_size,
                              hipStream_t stream) {
    const float* x_n1 = (const float*)d_in[0];
    const float* x_n2 = (const float*)d_in[1];
    const int* ei[3] = {(const int*)d_in[2], (const int*)d_in[3], (const int*)d_in[4]};
    const int* lei = (const int*)d_in[5];
    const int M1 = in_sizes[0] / HDIM;
    const int M2 = in_sizes[1] / HDIM;
    const int E = in_sizes[2] / 2;
    const int EL = in_sizes[5] / 2;

    // params flattened alphabetically: bn1(n1.b,n1.g,n2.b,n2.g), bn2(...),
    // l1(r1..r3: dst.W,dst.b,src.W,src.b,upd.W,upd.b), l2(...)
    const float* bn_b[2][2];
    const float* bn_g[2][2];
    bn_b[0][0] = (const float*)d_in[6];  bn_g[0][0] = (const float*)d_in[7];
    bn_b[0][1] = (const float*)d_in[8];  bn_g[0][1] = (const float*)d_in[9];
    bn_b[1][0] = (const float*)d_in[10]; bn_g[1][0] = (const float*)d_in[11];
    bn_b[1][1] = (const float*)d_in[12]; bn_g[1][1] = (const float*)d_in[13];

    // ---- workspace carve-out (256B aligned slabs)
    char* wsp = (char*)d_ws;
    auto take = [&](size_t b) {
        void* p = (void*)wsp;
        wsp += (b + 255) & ~(size_t)255;
        return p;
    };
    unsigned short* wf1[6];
    unsigned short* wf2[6];
    float* bfused[6];
    for (int c = 0; c < 6; ++c) {
        wf1[c] = (unsigned short*)take((size_t)HDIM * HDIM * 2);
        wf2[c] = (unsigned short*)take((size_t)HDIM * HDIM * 2);
        bfused[c] = (float*)take(HDIM * 4);
    }
    float* stats = (float*)take(512 * 4);
    unsigned short* hb1 = (unsigned short*)take((size_t)M1 * HDIM * 2);
    unsigned short* hb2 = (unsigned short*)take((size_t)M2 * HDIM * 2);
    int Mmax = M1 > M2 ? M1 : M2;
    float* aggsum = (float*)take((size_t)Mmax * HDIM * 4);
    unsigned short* aggb = (unsigned short*)take((size_t)Mmax * HDIM * 2);
    float* counts = (float*)take((size_t)Mmax * 4);
    float* e1 = (float*)take((size_t)M1 * HDIM * 4);
    float* e2 = (float*)take((size_t)M2 * HDIM * 4);

    // ---- input conversion + weight fusion
    int q1 = M1 * HDIM / 4, q2 = M2 * HDIM / 4;
    k_f32_to_bf16<<<(q1 + 255) / 256, 256, 0, stream>>>(x_n1, hb1, q1);
    k_f32_to_bf16<<<(q2 + 255) / 256, 256, 0, stream>>>(x_n2, hb2, q2);
    for (int c = 0; c < 6; ++c) {
        int base = 14 + (c / 3) * 18 + (c % 3) * 6;
        k_fuse_weights<<<HDIM, HDIM, 0, stream>>>(
            (const float*)d_in[base], (const float*)d_in[base + 1],
            (const float*)d_in[base + 2], (const float*)d_in[base + 3],
            (const float*)d_in[base + 4], (const float*)d_in[base + 5],
            wf1[c], wf2[c], bfused[c]);
    }

    auto conv = [&](int l, int r, const unsigned short* srcfeat,
                    const unsigned short* dstfeat, const int* eidx, int n_dst,
                    float* outp, int acc) {
        int c = l * 3 + r;
        (void)hipMemsetAsync(aggsum, 0, (size_t)n_dst * HDIM * 4, stream);
        (void)hipMemsetAsync(counts, 0, (size_t)n_dst * 4, stream);
        k_scatter_add<<<(E + 7) / 8, 256, 0, stream>>>(srcfeat, eidx, eidx + E,
                                                       aggsum, counts, E);
        int q = n_dst * HDIM / 4;
        k_mean_bf16<<<(q + 255) / 256, 256, 0, stream>>>(aggsum, counts, aggb, q);
        dim3 grid((n_dst / 16 + 7) / 8, HDIM / 16);
        k_conv_gemm<<<grid, 256, 0, stream>>>(dstfeat, aggb, wf1[c], wf2[c],
                                              bfused[c], outp, n_dst, acc);
    };
    auto bn = [&](const float* ebuf, float scale, const float* g, const float* bsh,
                  unsigned short* hb, int M) {
        (void)hipMemsetAsync(stats, 0, 256 * 4, stream);
        k_bn_stats<<<512, HDIM, 0, stream>>>(ebuf, scale, stats, M);
        k_bn_finalize<<<1, HDIM, 0, stream>>>(stats, M);
        int q = M * HDIM / 4;
        k_bn_apply<<<(q + 255) / 256, 256, 0, stream>>>(ebuf, scale, stats, g, bsh,
                                                        hb, q);
    };

    for (int l = 0; l < 2; ++l) {
        conv(l, 0, hb1, hb1, ei[0], M1, e1, 0);   // r1: n1 -> n1
        conv(l, 1, hb2, hb1, ei[1], M1, e1, 1);   // r2: n2 -> n1 (accumulate)
        conv(l, 2, hb1, hb2, ei[2], M2, e2, 0);   // r3: n1 -> n2
        bn(e1, 0.5f, bn_g[l][0], bn_b[l][0], hb1, M1);  // stack-mean folded in
        bn(e2, 1.0f, bn_g[l][1], bn_b[l][1], hb2, M2);
    }
    k_link_dot<<<(EL + 7) / 8, 256, 0, stream>>>(hb1, lei, lei + EL,
                                                 (float*)d_out, EL);
}
